// WindowAttention_25151328485413
// MI455X (gfx1250) — matrix operand order repaired and checked by reading, not compiled, not run on hardware
//
#include <hip/hip_runtime.h>
#include <hip/hip_bf16.h>
#include <cstdint>

typedef __attribute__((ext_vector_type(16))) __bf16 v16bf;
typedef __attribute__((ext_vector_type(8)))  __bf16 v8bf;
typedef __attribute__((ext_vector_type(4)))  __bf16 v4bf;
typedef __attribute__((ext_vector_type(8)))  float  v8f;

#define WSZ    7
#define NTOK   49          // 7*7 tokens per window
#define NWH    8           // windows per side (56/7)
#define NWIN   64
#define NHEADS 6
#define HD     32
#define CDIM   192
#define NB     32
#define NH     56
#define NWD    56
#define QKVC   576
#define NEGBIG (-1.0e9f)

__device__ __forceinline__ v8f wmma_bf16(v16bf a, v16bf b, v8f c) {
  return __builtin_amdgcn_wmma_f32_16x16x32_bf16(false, a, false, b, (short)0, c, false, false);
}
__device__ __forceinline__ v16bf cat16(v8bf lo, v8bf hi) {
  return __builtin_shufflevector(lo, hi, 0,1,2,3,4,5,6,7,8,9,10,11,12,13,14,15);
}
__device__ __forceinline__ v8bf ld8(const __bf16* p) { return *(const v8bf*)p; }

// ---- CDNA5 async copy (ASYNCcnt) fallback path --------------------------------
__device__ __forceinline__ void async_ld_b128(unsigned ldsoff, const void* sbase, int voff) {
  asm volatile("global_load_async_to_lds_b128 %0, %1, %2 offset:0"
               :: "v"(ldsoff), "v"(voff), "s"(sbase) : "memory");
}
__device__ __forceinline__ void wait_async0() {
  asm volatile("s_wait_asynccnt 0" ::: "memory");
}

// ---- CDNA5 Tensor Data Mover (TENSORcnt) ---------------------------------------
#if __has_builtin(__builtin_amdgcn_tensor_load_to_lds) && __has_builtin(__builtin_amdgcn_s_wait_tensorcnt)
#define HAVE_TDM 1
typedef __attribute__((ext_vector_type(4))) unsigned u32x4;
typedef __attribute__((ext_vector_type(8))) int      i32x8;
typedef __attribute__((ext_vector_type(4))) int      i32x4;

// 1-D global->LDS DMA expressed as an (N8 x 1) tile of 8-byte elements.
// D# bit layout per CDNA5 ISA (group0 / group1).
__device__ __forceinline__ void tdm_copy_1d(unsigned lds_off, const void* gsrc, unsigned nbytes) {
  const unsigned n8 = nbytes >> 3;
  const uint64_t ga = (uint64_t)(uintptr_t)gsrc;
  u32x4 g0;
  g0[0] = 1u;                                  // count=1, is_restore=0, gather off
  g0[1] = lds_off;                             // lds_addr (bytes)
  g0[2] = (unsigned)ga;                        // global_addr[31:0]
  g0[3] = (unsigned)(ga >> 32) | (2u << 30);   // global_addr[56:32] | type=2 (image)
  i32x8 g1;
  g1[0] = (int)(3u << 16);                     // data_size=3 (8B), mask=0, no flags
  g1[1] = (int)((n8 & 0xFFFFu) << 16);         // tensor_dim0[15:0] @ bit48
  g1[2] = (int)((n8 >> 16) | (1u << 16));      // tensor_dim0[31:16] | tensor_dim1=1
  g1[3] = (int)(n8 << 16);                     // tile_dim0 @ bit112
  g1[4] = 1;                                   // tile_dim1=1, tile_dim2=0
  g1[5] = (int)n8;                             // tensor_dim0_stride[31:0]
  g1[6] = 0;
  g1[7] = 0;
  i32x4 z = {0, 0, 0, 0};
#if __clang_major__ >= 23
  i32x8 z8 = {0, 0, 0, 0, 0, 0, 0, 0};
  __builtin_amdgcn_tensor_load_to_lds(g0, g1, z, z, z8, 0);
#else
  __builtin_amdgcn_tensor_load_to_lds(g0, g1, z, z, 0);
#endif
}
#endif

// ---------------------------------------------------------------------------
// Kernel 0a: one-shot weight transpose + f32->bf16 convert (L2-resident after).
// ---------------------------------------------------------------------------
__global__ __launch_bounds__(256) void wconv_kernel(const float* __restrict__ wqkv,
                                                    const float* __restrict__ wout,
                                                    __bf16* __restrict__ wTq,
                                                    __bf16* __restrict__ wTo) {
  int gid = blockIdx.x * 256 + threadIdx.x;
  if (gid < CDIM * QKVC) {
    int kk = gid / QKVC, col = gid - kk * QKVC;
    wTq[(size_t)col * CDIM + kk] = (__bf16)wqkv[gid];
  } else {
    int g2 = gid - CDIM * QKVC;
    if (g2 < CDIM * CDIM) {
      int kk = g2 / CDIM, col = g2 - kk * CDIM;
      wTo[(size_t)col * CDIM + kk] = (__bf16)wout[g2];
    }
  }
}

// ---------------------------------------------------------------------------
// Kernel 0b: precompute 4 padded 64x64 bias(+mask) tables:
// variant bit0 -> ul shift mask, bit1 -> lr shift mask. j>=49 -> NEG (softmax
// padding), i>=49 -> 0. Shared by all 12288 attention blocks (L2-resident).
// ---------------------------------------------------------------------------
__global__ __launch_bounds__(256) void bias_kernel(const float* __restrict__ pos_emb,
                                                   float* __restrict__ btab) {
  int gid = blockIdx.x * 256 + threadIdx.x;    // variant*4096 + i*64 + j
  int variant = gid >> 12;
  int rem = gid & 4095;
  int i = rem >> 6, j = rem & 63;
  float v;
  if (j >= NTOK)      v = NEGBIG;
  else if (i >= NTOK) v = 0.0f;
  else {
    int xi = i / WSZ, yi = i - xi * WSZ;
    int xj = j / WSZ, yj = j - xj * WSZ;
    v = pos_emb[(xj - xi + WSZ - 1) * (2 * WSZ - 1) + (yj - yi + WSZ - 1)];
    if ((variant & 1) && ((i >= 28) != (j >= 28))) v += NEGBIG;   // ul mask
    if ((variant & 2) && ((yi >= 4) != (yj >= 4))) v += NEGBIG;   // lr mask
  }
  btab[gid] = v;
}

// ---------------------------------------------------------------------------
// Kernel 1: rolled x -> QKV GEMM (bf16 WMMA). Scatter path is wave-uniform:
// qsel/head derived from nt (scalar branch), per-row offsets hoisted as ints.
// ---------------------------------------------------------------------------
__global__ __launch_bounds__(256) void qkv_kernel(const float* __restrict__ x,
                                                  const __bf16* __restrict__ wTq,
                                                  __bf16* __restrict__ q,
                                                  __bf16* __restrict__ k,
                                                  __bf16* __restrict__ vt) {
  __shared__ alignas(16) __bf16 ldsA[16 * CDIM];
  const int mtile = blockIdx.x;
  const int tid   = threadIdx.x;

  for (int idx = tid; idx < 16 * (CDIM / 4); idx += 256) {
    int r = idx / (CDIM / 4), c4 = (idx - r * (CDIM / 4)) * 4;
    int t   = mtile * 16 + r;
    int bb  = t / (NH * NWD);
    int rem = t - bb * NH * NWD;
    int hh  = rem / NWD, ww = rem - hh * NWD;
    int hs = hh + 3; if (hs >= NH)  hs -= NH;
    int ws = ww + 3; if (ws >= NWD) ws -= NWD;
    const float4 f = *(const float4*)(x + ((size_t)(bb * NH + hs) * NWD + ws) * CDIM + c4);
    v4bf o = { (__bf16)f.x, (__bf16)f.y, (__bf16)f.z, (__bf16)f.w };
    *(v4bf*)&ldsA[r * CDIM + c4] = o;
  }
  __syncthreads();

  const int lane = tid & 31, wave = tid >> 5, hi = lane >> 4, lrow = lane & 15;

  v16bf afrag[6];
#pragma unroll
  for (int kk = 0; kk < 6; ++kk) {
    const __bf16* p = &ldsA[lrow * CDIM + kk * 32 + hi * 8];
    afrag[kk] = cat16(ld8(p), ld8(p + 16));
  }

  int roff[8], vroff[8];
#pragma unroll
  for (int r = 0; r < 8; ++r) {
    int t   = mtile * 16 + r + hi * 8;
    int bb  = t / (NH * NWD);
    int rem = t - bb * NH * NWD;
    int hh  = rem / NWD, ww = rem - hh * NWD;
    int win = (hh / WSZ) * NWH + (ww / WSZ);
    int n   = (hh % WSZ) * WSZ + (ww % WSZ);
    roff[r]  = bb * (NHEADS * NWIN * NTOK * HD) + win * (NTOK * HD) + n * HD;
    vroff[r] = bb * (NHEADS * NWIN * HD * 64)  + win * (HD * 64)  + n;
  }

  for (int nt = wave; nt < QKVC / 16; nt += 8) {
    const int col = nt * 16 + lrow;
    const __bf16* bp = wTq + (size_t)col * CDIM;
    __builtin_prefetch(bp, 0, 3);                 // global_prefetch_b8
    v8f acc = {};
#pragma unroll
    for (int kk = 0; kk < 6; ++kk) {
      v16bf bm = cat16(ld8(bp + kk * 32 + hi * 8), ld8(bp + kk * 32 + hi * 8 + 16));
      acc = wmma_bf16(afrag[kk], bm, acc);
    }
    const int colbase = nt * 16;
    const int qsel  = colbase / CDIM;
    const int c2b   = colbase - qsel * CDIM;
    const int head  = c2b >> 5;
    const int d     = (c2b & 31) + lrow;
    if (qsel == 0) {
      const int hb = head * (NWIN * NTOK * HD) + d;
#pragma unroll
      for (int r = 0; r < 8; ++r) q[roff[r] + hb] = (__bf16)acc[r];
    } else if (qsel == 1) {
      const int hb = head * (NWIN * NTOK * HD) + d;
#pragma unroll
      for (int r = 0; r < 8; ++r) k[roff[r] + hb] = (__bf16)acc[r];
    } else {
      const int hb = head * (NWIN * HD * 64) + d * 64;
#pragma unroll
      for (int r = 0; r < 8; ++r) vt[vroff[r] + hb] = (__bf16)acc[r];   // transposed
    }
  }
}

// ---------------------------------------------------------------------------
// Kernel 2: per-(b,head,window) attention. q/k/vT + bias table DMA'd into LDS
// via TDM (TENSORcnt) when available, else async-to-LDS (ASYNCcnt).
// Scoring = one WMMA + one ds_load + one fma per element; shfl softmax.
// ---------------------------------------------------------------------------
__global__ __launch_bounds__(128) void attn_kernel(const __bf16* __restrict__ qg,
                                                   const __bf16* __restrict__ kg,
                                                   const __bf16* __restrict__ vtg,
                                                   const float* __restrict__ btab,
                                                   __bf16* __restrict__ ao) {
  __shared__ alignas(16) __bf16 lq[NTOK * HD];
  __shared__ alignas(16) __bf16 lk[NTOK * HD];
  __shared__ alignas(16) __bf16 lv[HD * 64];
  __shared__ alignas(16) __bf16 lattn[4 * 16 * 64];
  __shared__ alignas(16) float  lbias[64 * 64];

  const int idx  = blockIdx.x;               // (bb*6+head)*64 + win
  const int win  = idx & 63, bh = idx >> 6;
  const int head = bh % NHEADS, bb = bh / NHEADS;
  const int wrow = win >> 3, wcol = win & 7;
  const int tid  = threadIdx.x;
  const int lane = tid & 31, wave = tid >> 5, hi = lane >> 4, lrow = lane & 15;
  const float scale = 0.17677669529663687f;  // 1/sqrt(32)

  const __bf16* gq = qg  + (size_t)idx * (NTOK * HD);
  const __bf16* gk = kg  + (size_t)idx * (NTOK * HD);
  const __bf16* gv = vtg + (size_t)idx * (HD * 64);
  const int variant = (wrow == NWH - 1 ? 1 : 0) | (wcol == NWH - 1 ? 2 : 0);
  const float* gb = btab + variant * 4096;   // block-uniform table select

#ifdef HAVE_TDM
  if (tid < 32) {                            // one wave issues the 4 descriptors
    tdm_copy_1d((unsigned)(uintptr_t)(void*)lq,    gq, NTOK * HD * 2);
    tdm_copy_1d((unsigned)(uintptr_t)(void*)lk,    gk, NTOK * HD * 2);
    tdm_copy_1d((unsigned)(uintptr_t)(void*)lv,    gv, HD * 64 * 2);
    tdm_copy_1d((unsigned)(uintptr_t)(void*)lbias, gb, 64 * 64 * 4);
    __builtin_amdgcn_s_wait_tensorcnt(0);
  }
#else
  {
    unsigned oq = (unsigned)(uintptr_t)(void*)lq;
    unsigned ok = (unsigned)(uintptr_t)(void*)lk;
    unsigned ov = (unsigned)(uintptr_t)(void*)lv;
    unsigned ob = (unsigned)(uintptr_t)(void*)lbias;
    for (int i = tid; i < (NTOK * HD) / 8; i += 128) {
      async_ld_b128(oq + i * 16, gq, i * 16);
      async_ld_b128(ok + i * 16, gk, i * 16);
    }
    for (int i = tid; i < (HD * 64) / 8; i += 128)
      async_ld_b128(ov + i * 16, gv, i * 16);
    for (int i = tid; i < (64 * 64) / 4; i += 128)
      async_ld_b128(ob + i * 16, gb, i * 16);
    wait_async0();
  }
#endif
  __syncthreads();

  // ---- dots = q @ k^T, scored via precomputed table: fma only ----
  const int irow = wave * 16 + lrow;
  const int qr   = irow < NTOK ? irow : NTOK - 1;
  const __bf16* apq = &lq[qr * HD + hi * 8];
  v16bf aq = cat16(ld8(apq), ld8(apq + 16));

  const float* brow = &lbias[(wave * 16 + hi * 8) * 64 + lrow];  // + r*64 + nt*16
  float vals[4][8];
#pragma unroll
  for (int nt = 0; nt < 4; ++nt) {
    const int j  = nt * 16 + lrow;
    const int jr = j < NTOK ? j : NTOK - 1;
    const __bf16* kp = &lk[jr * HD + hi * 8];
    v16bf bk = cat16(ld8(kp), ld8(kp + 16));
    v8f c = {};
    c = wmma_bf16(aq, bk, c);
#pragma unroll
    for (int r = 0; r < 8; ++r)
      vals[nt][r] = fmaf(c[r], scale, brow[r * 64 + nt * 16]);
  }

  // ---- softmax ----
#pragma unroll
  for (int r = 0; r < 8; ++r) {
    float m = vals[0][r];
#pragma unroll
    for (int nt = 1; nt < 4; ++nt) m = fmaxf(m, vals[nt][r]);
#pragma unroll
    for (int off = 8; off > 0; off >>= 1) m = fmaxf(m, __shfl_xor(m, off, 16));
    float s = 0.0f;
#pragma unroll
    for (int nt = 0; nt < 4; ++nt) { vals[nt][r] = __expf(vals[nt][r] - m); s += vals[nt][r]; }
#pragma unroll
    for (int off = 8; off > 0; off >>= 1) s += __shfl_xor(s, off, 16);
    float inv = 1.0f / s;
#pragma unroll
    for (int nt = 0; nt < 4; ++nt) vals[nt][r] *= inv;
  }

  // ---- transpose attn into WMMA A-layout via LDS ----
#pragma unroll
  for (int nt = 0; nt < 4; ++nt)
#pragma unroll
    for (int r = 0; r < 8; ++r)
      lattn[wave * 1024 + (r + hi * 8) * 64 + nt * 16 + lrow] = (__bf16)vals[nt][r];
  __syncthreads();

  const __bf16* tp = &lattn[wave * 1024 + lrow * 64];
  v16bf a0 = cat16(ld8(tp +      hi * 8), ld8(tp + 16 + hi * 8));
  v16bf a1 = cat16(ld8(tp + 32 + hi * 8), ld8(tp + 48 + hi * 8));

  int aoff[8];
#pragma unroll
  for (int r = 0; r < 8; ++r) {
    int tok = wave * 16 + r + hi * 8;
    if (tok < NTOK) {
      int hh = wrow * WSZ + tok / WSZ;
      int ww = wcol * WSZ + tok % WSZ;
      aoff[r] = ((bb * NH + hh) * NWD + ww) * CDIM + head * HD;
    } else aoff[r] = -1;
  }

  // ---- out = attn @ v ----
#pragma unroll
  for (int nt2 = 0; nt2 < 2; ++nt2) {
    const int n = nt2 * 16 + lrow;
    const __bf16* vp = &lv[n * 64];
    v16bf b0 = cat16(ld8(vp +      hi * 8), ld8(vp +      hi * 8 + 16));
    v16bf b1 = cat16(ld8(vp + 32 + hi * 8), ld8(vp + 32 + hi * 8 + 16));
    v8f oc = {};
    oc = wmma_bf16(a0, b0, oc);
    oc = wmma_bf16(a1, b1, oc);
#pragma unroll
    for (int r = 0; r < 8; ++r)
      if (aoff[r] >= 0) ao[(size_t)aoff[r] + n] = (__bf16)oc[r];
  }
}

// ---------------------------------------------------------------------------
// Kernel 3: out projection (bf16 WMMA) + bias, roll(+3,+3) folded into store.
// ---------------------------------------------------------------------------
__global__ __launch_bounds__(256) void proj_kernel(const __bf16* __restrict__ ao,
                                                   const __bf16* __restrict__ wTo,
                                                   const float* __restrict__ bout,
                                                   float* __restrict__ out) {
  const int mtile = blockIdx.x, tid = threadIdx.x;
  const int lane = tid & 31, wave = tid >> 5, hi = lane >> 4, lrow = lane & 15;

  const __bf16* arow = ao + (size_t)(mtile * 16 + lrow) * CDIM;
  v16bf afrag[6];
#pragma unroll
  for (int kk = 0; kk < 6; ++kk)
    afrag[kk] = cat16(ld8(arow + kk * 32 + hi * 8), ld8(arow + kk * 32 + 16 + hi * 8));

  int dstoff[8];
#pragma unroll
  for (int r = 0; r < 8; ++r) {
    int t   = mtile * 16 + r + hi * 8;
    int bb  = t / (NH * NWD);
    int rem = t - bb * NH * NWD;
    int hh  = rem / NWD, ww = rem - hh * NWD;
    int h2 = hh + 3; if (h2 >= NH)  h2 -= NH;
    int w2 = ww + 3; if (w2 >= NWD) w2 -= NWD;
    dstoff[r] = ((bb * NH + h2) * NWD + w2) * CDIM;
  }

  for (int nt = wave; nt < CDIM / 16; nt += 8) {
    const int col = nt * 16 + lrow;
    const __bf16* bp = wTo + (size_t)col * CDIM;
    __builtin_prefetch(bp, 0, 3);
    v8f acc = {};
#pragma unroll
    for (int kk = 0; kk < 6; ++kk) {
      v16bf bm = cat16(ld8(bp + kk * 32 + hi * 8), ld8(bp + kk * 32 + hi * 8 + 16));
      acc = wmma_bf16(afrag[kk], bm, acc);
    }
    const float bias = bout[col];
#pragma unroll
    for (int r = 0; r < 8; ++r)
      out[(size_t)dstoff[r] + col] = acc[r] + bias;
  }
}

// ---------------------------------------------------------------------------
extern "C" void kernel_launch(void* const* d_in, const int* in_sizes, int n_in,
                              void* d_out, int out_size, void* d_ws, size_t ws_size,
                              hipStream_t stream) {
  (void)in_sizes; (void)n_in; (void)out_size; (void)ws_size;
  const float* x       = (const float*)d_in[0];
  const float* w_qkv   = (const float*)d_in[1];
  const float* w_out   = (const float*)d_in[2];
  const float* b_out   = (const float*)d_in[3];
  const float* pos_emb = (const float*)d_in[4];

  const size_t QS  = (size_t)NB * NHEADS * NWIN * NTOK * HD;   // q/k elems
  const size_t VTS = (size_t)NB * NHEADS * NWIN * HD * 64;     // padded vT elems
  __bf16* wsq  = (__bf16*)d_ws;
  __bf16* wsk  = wsq  + QS;
  __bf16* wsvT = wsk  + QS;
  __bf16* wsao = wsvT + VTS;
  __bf16* wTq  = wsao + QS;
  __bf16* wTo  = wTq  + (size_t)CDIM * QKVC;
  float*  btab = (float*)(wTo + (size_t)CDIM * CDIM);          // 4 x 64x64 tables

  const int mtiles = (NB * NH * NWD) / 16;                     // 6272
  wconv_kernel<<<dim3((CDIM * QKVC + CDIM * CDIM) / 256), 256, 0, stream>>>(w_qkv, w_out, wTq, wTo);
  bias_kernel <<<dim3(64), 256, 0, stream>>>(pos_emb, btab);
  hipMemsetAsync(wsvT, 0, VTS * sizeof(__bf16), stream);
  qkv_kernel <<<dim3(mtiles), 256, 0, stream>>>(x, wTq, wsq, wsk, wsvT);
  attn_kernel<<<dim3(NB * NHEADS * NWIN), 128, 0, stream>>>(wsq, wsk, wsvT, btab, wsao);
  proj_kernel<<<dim3(mtiles), 256, 0, stream>>>(wsao, wTo, b_out, (float*)d_out);
}
